// CrossAttentionReducer_2147483648354
// MI455X (gfx1250) — matrix operand order repaired and checked by reading, not compiled, not run on hardware
//
#include <hip/hip_runtime.h>
#include <hip/hip_bf16.h>
#include <cstdint>

typedef __attribute__((ext_vector_type(16))) _Float16 v16h;
typedef __attribute__((ext_vector_type(8)))  _Float16 v8h;
typedef __attribute__((ext_vector_type(8)))  float    v8f;
typedef __attribute__((ext_vector_type(4)))  float    v4f;

#define LLM_DIM 4096
#define VOCAB   4096
#define HEADS   8
#define INNER   512
#define HEAD_DIM 64
#define SEQ_T   8192

__device__ __forceinline__ v8f wmma_f16(v16h a, v16h b, v8f c) {
    // D = A(16x32 f16) * B(32x16 f16) + C(16x16 f32)
    return __builtin_amdgcn_wmma_f32_16x16x32_f16(
        /*neg_a=*/false, a, /*neg_b=*/false, b,
        /*c_mod=*/(short)0, c, /*reuse_a=*/false, /*reuse_b=*/false);
}

__device__ __forceinline__ v16h cat8(v8h a, v8h b) {
    return __builtin_shufflevector(a, b, 0,1,2,3,4,5,6,7,8,9,10,11,12,13,14,15);
}

// Async global->LDS copy (16 bytes), tracked by ASYNCcnt; bypasses VGPRs.
// The generic pointer to a __shared__ object carries the 32-bit LDS byte
// offset in its low dword (addrspacecast AS3->generic = {aperture, offset}).
__device__ __forceinline__ void async_copy_b128(void* lds_dst, const void* gsrc) {
    uint32_t lds_off = (uint32_t)(uintptr_t)lds_dst;
    asm volatile("global_load_async_to_lds_b128 %0, %1, off"
                 :: "v"(lds_off), "v"(gsrc) : "memory");
}
__device__ __forceinline__ void wait_async0() {
    asm volatile("s_wait_asynccnt 0x0" ::: "memory");
}

// ---------------------------------------------------------------------------
// Tiled GEMM: C[M,N] = A[M,K] * B[K,N]
//   A: row-major fp32 (A_F16=false, convert on LDS path) or f16 (async copy)
//   B: row-major fp32, converted to f16, staged transposed [n][k]
//   C: f16 (OUT_F16=true) or fp32
// Block: 256 threads (8 waves), BM=128 BN=128 BK=32.
// Wave grid 4x2: each wave computes a 32x64 region = 2x4 WMMA tiles.
// ---------------------------------------------------------------------------
constexpr int BM = 128, BN = 128, BK = 32;

template<bool A_F16, bool OUT_F16>
__global__ __launch_bounds__(256)
void gemm_wmma_kernel(const void* __restrict__ Aptr,
                      const float* __restrict__ B,
                      void* __restrict__ Cptr,
                      int M, int N, int K)
{
    (void)M;
    __shared__ __align__(32) _Float16 As[BM][BK];   // 8 KB, row-major [m][k]
    __shared__ __align__(32) _Float16 Bt[BN][BK];   // 8 KB, transposed [n][k]

    const int tid  = threadIdx.x;
    const int lane = tid & 31;
    const int wave = tid >> 5;       // 0..7
    const int wm   = wave >> 1;      // 0..3
    const int wn   = wave & 1;       // 0..1
    const int half = lane >> 4;      // 0/1
    const int l16  = lane & 15;
    const int m0   = blockIdx.y * BM;
    const int n0   = blockIdx.x * BN;

    const v8f zf = {0.f,0.f,0.f,0.f,0.f,0.f,0.f,0.f};
    v8f acc[2][4];
    #pragma unroll
    for (int i = 0; i < 2; ++i)
        #pragma unroll
        for (int j = 0; j < 4; ++j) acc[i][j] = zf;

    for (int k0 = 0; k0 < K; k0 += BK) {
        __syncthreads();
        // ---- stage A tile: 128x32, each thread handles 16 contiguous k ----
        {
            const int row = tid >> 1;
            const int cb  = (tid & 1) * 16;
            if (A_F16) {
                const _Float16* A = (const _Float16*)Aptr;
                const _Float16* src = A + (size_t)(m0 + row) * K + k0 + cb;
                // pure byte copy -> async DMA path, no VGPR round-trip
                async_copy_b128(&As[row][cb],     src);
                async_copy_b128(&As[row][cb + 8], src + 8);
            } else {
                const float* A = (const float*)Aptr;
                const float* src = A + (size_t)(m0 + row) * K + k0 + cb;
                v4f f0 = *(const v4f*)(src + 0);
                v4f f1 = *(const v4f*)(src + 4);
                v4f f2 = *(const v4f*)(src + 8);
                v4f f3 = *(const v4f*)(src + 12);
                v8h h0, h1;
                #pragma unroll
                for (int i = 0; i < 4; ++i) {
                    h0[i]   = (_Float16)f0[i];
                    h0[4+i] = (_Float16)f1[i];
                    h1[i]   = (_Float16)f2[i];
                    h1[4+i] = (_Float16)f3[i];
                }
                *(v8h*)&As[row][cb]     = h0;
                *(v8h*)&As[row][cb + 8] = h1;
            }
        }
        // ---- stage B tile transposed: Bt[n][k], each thread 16 k's ----
        {
            const int n  = tid >> 1;
            const int kb = (tid & 1) * 16;
            // prefetch the next k-step's column walk into L2
            if (k0 + BK < K)
                __builtin_prefetch(&B[(size_t)(k0 + BK + kb) * N + n0 + n], 0, 1);
            #pragma unroll
            for (int i = 0; i < 16; ++i)
                Bt[n][kb + i] = (_Float16)B[(size_t)(k0 + kb + i) * N + n0 + n];
        }
        if (A_F16) wait_async0();     // async LDS writes visible before barrier
        __syncthreads();

        // ---- fragments + WMMA ----
        v16h afrag[2], bfrag[4];
        #pragma unroll
        for (int mt = 0; mt < 2; ++mt) {
            const int row = wm * 32 + mt * 16 + l16;        // A: m = lane%16
            v8h a0 = *(const v8h*)&As[row][half * 8];       // k = half*8 + 0..7
            v8h a1 = *(const v8h*)&As[row][16 + half * 8];  // k = 16+half*8 + 0..7
            afrag[mt] = cat8(a0, a1);
        }
        #pragma unroll
        for (int nt = 0; nt < 4; ++nt) {
            const int col = wn * 64 + nt * 16 + l16;        // B: n = lane%16
            bfrag[nt] = cat8(*(const v8h*)&Bt[col][half * 8], *(const v8h*)&Bt[col][16 + half * 8]);  // k = half*8 + 0..7, 16+half*8 + 0..7
        }
        #pragma unroll
        for (int mt = 0; mt < 2; ++mt)
            #pragma unroll
            for (int nt = 0; nt < 4; ++nt)
                acc[mt][nt] = wmma_f16(afrag[mt], bfrag[nt], acc[mt][nt]);
    }

    // ---- epilogue: D tile rows m = (lane/16)*8 + r ----
    #pragma unroll
    for (int mt = 0; mt < 2; ++mt) {
        #pragma unroll
        for (int nt = 0; nt < 4; ++nt) {
            const int gm0 = m0 + wm * 32 + mt * 16 + half * 8;
            const int gn  = n0 + wn * 64 + nt * 16 + l16;
            v8f c = acc[mt][nt];
            if (OUT_F16) {
                _Float16* C = (_Float16*)Cptr;
                #pragma unroll
                for (int r = 0; r < 8; ++r)
                    C[(size_t)(gm0 + r) * N + gn] = (_Float16)c[r];
            } else {
                float* C = (float*)Cptr;
                #pragma unroll
                for (int r = 0; r < 8; ++r)
                    C[(size_t)(gm0 + r) * N + gn] = c[r];
            }
        }
    }
}

// ---------------------------------------------------------------------------
// Fused attention: O[v, h*64+d] = softmax_t( Q·Kt / 8 ) · V, per head.
// Grid: (VOCAB/128, HEADS). Block 256 threads = 8 waves; wave owns 16 q-rows.
// Single-pass exp-sum softmax (scores are O(1) here so no running max needed).
// K tile staged via async global->LDS DMA; V tile transposed through VGPRs.
// ---------------------------------------------------------------------------
__global__ __launch_bounds__(256)
void attn_kernel(const _Float16* __restrict__ Q,   // [VOCAB, INNER] f16
                 const _Float16* __restrict__ Km,  // [T, INNER] f16
                 const _Float16* __restrict__ Vm,  // [T, INNER] f16
                 _Float16* __restrict__ O)         // [VOCAB, INNER] f16
{
    constexpr int BT = 64;
    __shared__ __align__(32) _Float16 Kt[BT][HEAD_DIM];   // [t][d] row-major
    __shared__ __align__(32) _Float16 Vt[HEAD_DIM][BT];   // [d][t] transposed
    __shared__ __align__(32) _Float16 Pt[8][16][BT];      // per-wave P tiles

    const int tid  = threadIdx.x;
    const int lane = tid & 31;
    const int wave = tid >> 5;
    const int half = lane >> 4;
    const int l16  = lane & 15;
    const int h    = blockIdx.y;
    const int hc   = h * HEAD_DIM;
    const int q0   = blockIdx.x * 128 + wave * 16;

    // preload Q fragments for the d-contraction (2 k-steps of 32)
    v16h qfrag[2];
    {
        const _Float16* qrow = Q + (size_t)(q0 + l16) * INNER + hc;
        #pragma unroll
        for (int ks = 0; ks < 2; ++ks) {
            v8h a0 = *(const v8h*)(qrow + ks * 32 + half * 8);
            v8h a1 = *(const v8h*)(qrow + ks * 32 + 16 + half * 8);
            qfrag[ks] = cat8(a0, a1);
        }
    }

    const v8f zf = {0.f,0.f,0.f,0.f,0.f,0.f,0.f,0.f};
    v8f o_acc[4] = {zf, zf, zf, zf};
    float rowsum[8] = {0.f,0.f,0.f,0.f,0.f,0.f,0.f,0.f};
    const float scale = 0.125f;  // 1/sqrt(64)

    for (int t0 = 0; t0 < SEQ_T; t0 += BT) {
        __syncthreads();
        // stage K tile [t][d] via async DMA (layout-preserving byte copy)
        {
            const int tt = tid >> 2;
            const int db = (tid & 3) * 16;
            const _Float16* src = Km + (size_t)(t0 + tt) * INNER + hc + db;
            async_copy_b128(&Kt[tt][db],     src);
            async_copy_b128(&Kt[tt][db + 8], src + 8);
        }
        // stage V tile transposed [d][t] (B-frag layout for P·V)
        {
            const int tt = tid >> 2;
            const int db = (tid & 3) * 16;
            const _Float16* src = Vm + (size_t)(t0 + tt) * INNER + hc + db;
            v8h v0 = *(const v8h*)(src);
            v8h v1 = *(const v8h*)(src + 8);
            #pragma unroll
            for (int i = 0; i < 8; ++i) {
                Vt[db + i][tt]     = v0[i];
                Vt[db + 8 + i][tt] = v1[i];
            }
        }
        wait_async0();               // K-tile async writes visible in LDS
        __syncthreads();

        // S = Q·Kt, exp, row-sum, stage P into per-wave LDS
        #pragma unroll
        for (int nt = 0; nt < 4; ++nt) {
            v8f s = zf;
            #pragma unroll
            for (int ks = 0; ks < 2; ++ks) {
                v16h b = cat8(*(const v8h*)&Kt[nt * 16 + l16][ks * 32 + half * 8], *(const v8h*)&Kt[nt * 16 + l16][ks * 32 + 16 + half * 8]);
                s = wmma_f16(qfrag[ks], b, s);
            }
            #pragma unroll
            for (int r = 0; r < 8; ++r) {
                float e = __expf(s[r] * scale);
                // reduce over the 16 lanes sharing this row (n = 0..15)
                float v = e;
                v += __shfl_xor(v, 1);
                v += __shfl_xor(v, 2);
                v += __shfl_xor(v, 4);
                v += __shfl_xor(v, 8);
                rowsum[r] += v;
                Pt[wave][half * 8 + r][nt * 16 + l16] = (_Float16)e;
            }
        }

        // O += P·V (per-wave LDS round-trip converts D layout -> A layout)
        #pragma unroll
        for (int dt = 0; dt < 4; ++dt) {
            #pragma unroll
            for (int ks = 0; ks < 2; ++ks) {
                v8h p0 = *(const v8h*)&Pt[wave][l16][ks * 32 + half * 8];
                v8h p1 = *(const v8h*)&Pt[wave][l16][ks * 32 + 16 + half * 8];
                v16h a = cat8(p0, p1);
                v16h b = cat8(*(const v8h*)&Vt[dt * 16 + l16][ks * 32 + half * 8], *(const v8h*)&Vt[dt * 16 + l16][ks * 32 + 16 + half * 8]);
                o_acc[dt] = wmma_f16(a, b, o_acc[dt]);
            }
        }
    }

    // normalize and store (rows m = half*8 + r match rowsum[r] on this lane)
    #pragma unroll
    for (int dt = 0; dt < 4; ++dt) {
        #pragma unroll
        for (int r = 0; r < 8; ++r) {
            const float val = o_acc[dt][r] / rowsum[r];
            const int m = q0 + half * 8 + r;
            O[(size_t)m * INNER + hc + dt * 16 + l16] = (_Float16)val;
        }
    }
}

// ---------------------------------------------------------------------------
extern "C" void kernel_launch(void* const* d_in, const int* in_sizes, int n_in,
                              void* d_out, int out_size, void* d_ws, size_t ws_size,
                              hipStream_t stream) {
    (void)in_sizes; (void)n_in; (void)out_size; (void)ws_size;
    const float* tok  = (const float*)d_in[0];   // [T, LLM_DIM]
    const float* lq   = (const float*)d_in[1];   // [VOCAB, INNER]
    const float* wq   = (const float*)d_in[2];   // [INNER, INNER]
    const float* wk   = (const float*)d_in[3];   // [LLM_DIM, INNER]
    const float* wv   = (const float*)d_in[4];   // [LLM_DIM, INNER]
    const float* wout = (const float*)d_in[5];   // [INNER, LLM_DIM]
    float* out = (float*)d_out;                  // [VOCAB, LLM_DIM]

    _Float16* q16 = (_Float16*)d_ws;                        // 4096*512
    _Float16* k16 = q16 + (size_t)VOCAB * INNER;            // 8192*512
    _Float16* v16 = k16 + (size_t)SEQ_T * INNER;            // 8192*512
    _Float16* a16 = v16 + (size_t)SEQ_T * INNER;            // 4096*512

    dim3 blk(256);
    // q = lq @ w_q        [4096 x 512], K=512
    gemm_wmma_kernel<false, true>
        <<<dim3(INNER / BN, VOCAB / BM), blk, 0, stream>>>(lq, wq, q16, VOCAB, INNER, INNER);
    // k = tok @ w_k       [8192 x 512], K=4096
    gemm_wmma_kernel<false, true>
        <<<dim3(INNER / BN, SEQ_T / BM), blk, 0, stream>>>(tok, wk, k16, SEQ_T, INNER, LLM_DIM);
    // v = tok @ w_v
    gemm_wmma_kernel<false, true>
        <<<dim3(INNER / BN, SEQ_T / BM), blk, 0, stream>>>(tok, wv, v16, SEQ_T, INNER, LLM_DIM);
    // fused attention -> a16 [4096 x 512]
    attn_kernel<<<dim3(VOCAB / 128, HEADS), blk, 0, stream>>>(q16, k16, v16, a16);
    // out = a16 @ w_out   [4096 x 4096], K=512, fp32 output
    gemm_wmma_kernel<true, false>
        <<<dim3(LLM_DIM / BN, VOCAB / BM), blk, 0, stream>>>(a16, wout, out, VOCAB, LLM_DIM, INNER);
}
